// LilletLayer_68118181314881
// MI455X (gfx1250) — compile-verified, matrix operand order checked by reading, not run on hardware
//
#include <hip/hip_runtime.h>
#include <stdint.h>

// ---------------------------------------------------------------------------
// MI455X (gfx1250) implementation.
//  k0 pack_w   : permute [Wl;Wr] (256x3200 f32) into bf16 A-fragment-swizzled
//                panel (f' = r*64 + p ordering; f = p*50 + r in source).
//  k0b pack_x  : x (8x2048x3 f32) into bf16 B-fragment panel for map_xc.
//  k1 map_xc   : xc = W_map(8192x2048) x X(2048x24) via bf16 WMMA; 2-way
//                K-split per row-tile + LDS reduce (HBM-bound: 64 MiB).
//  k2 fused_att: per-WG 32 (b,g) columns; TDM loads xc tile to LDS; stage
//                basis bf16 B-fragments in LDS (double buffered),
//                16x16x32 bf16 WMMA over K=3200, lane-local sum_d L*R -> att.
//  k3 pool_mlp : logsumexp over g + 2-layer MLP.
// ---------------------------------------------------------------------------

typedef __attribute__((ext_vector_type(16))) __bf16 v16bf;
typedef __attribute__((ext_vector_type(8)))  float  v8f;

#define F_TOT   3200     // RBF*M*M
#define KCHUNKS 100      // 3200/32
#define BGTILE  32       // (b,g) columns per workgroup in k2
#define SRCH    10       // WMMA k-chunks staged per phase (320 k = 5 full r)

#define USE_TDM 1
#if USE_TDM && __has_builtin(__builtin_amdgcn_tensor_load_to_lds) && \
    __has_builtin(__builtin_amdgcn_s_wait_tensorcnt)
#define HAVE_TDM 1
typedef __attribute__((ext_vector_type(4))) unsigned int v4u;
typedef __attribute__((ext_vector_type(8))) int v8i;
typedef __attribute__((ext_vector_type(4))) int v4i;
#endif

__device__ __forceinline__ uint32_t pack_bf2(float a, float b) {
#if __has_builtin(__builtin_amdgcn_cvt_pk_bf16_f32)
  auto v = __builtin_amdgcn_cvt_pk_bf16_f32(a, b);   // single v_cvt_pk_bf16_f32
  return __builtin_bit_cast(uint32_t, v);
#else
  __bf16 lo = (__bf16)a, hi = (__bf16)b;
  unsigned short ul = __builtin_bit_cast(unsigned short, lo);
  unsigned short uh = __builtin_bit_cast(unsigned short, hi);
  return (uint32_t)ul | ((uint32_t)uh << 16);
#endif
}

union FragBF { uint32_t u[8]; uint4 q[2]; v16bf f; };

// A/B 16-bit fragment k-swizzle (wave32): lane<16 holds K {0..7,16..23},
// lane>=16 holds K {8..15,24..31}; VGPR v, element e: see ISA 7.12.2.
__device__ __forceinline__ int frag_k(int v, int half) {
  return ((v >= 4) ? 16 : 0) + half * 8 + (v & 3) * 2;
}

#define WMMA_BF16(A, Bv, C) \
  __builtin_amdgcn_wmma_f32_16x16x32_bf16(false, (A), false, (Bv), (short)0, (C), false, false)

// ---------------------------------------------------------------------------
// k0: pack [W_left; W_right] into bf16, fragment-swizzled, f'=r*64+p order.
// Word layout: [rt(16)][kc(100)][lane(32)*8 + v]  (matches k2 A loads).
// ---------------------------------------------------------------------------
__global__ __launch_bounds__(256) void pack_w(const float* __restrict__ Wl,
                                              const float* __restrict__ Wr,
                                              uint32_t* __restrict__ Apk) {
  int widx = blockIdx.x * 256 + threadIdx.x;      // 16*100*256 = 409600
  if (widx >= 16 * KCHUNKS * 256) return;
  int v    = widx & 7;
  int lane = (widx >> 3) & 31;
  int kc   = (widx >> 8) % KCHUNKS;
  int rt   = (widx >> 8) / KCHUNKS;
  int k0   = kc * 32 + frag_k(v, lane >> 4);      // f' index (even)
  int row  = rt * 16 + (lane & 15);               // 0..255 : [Wl;Wr]
  int p0 = k0 & 63, r0 = k0 >> 6;
  int p1 = (k0 + 1) & 63, r1 = (k0 + 1) >> 6;
  int f0 = p0 * 50 + r0;                          // inverse permutation
  int f1 = p1 * 50 + r1;
  const float* W = (row < 128) ? Wl : Wr;
  int rr = row & 127;
  Apk[widx] = pack_bf2(W[rr * F_TOT + f0], W[rr * F_TOT + f1]);
}

// ---------------------------------------------------------------------------
// k0b: pack x into bf16 B-fragment panel. Layout: [kc(64)][ct(2)][lane*8+v].
// Column ct*16+(lane&15) -> (b,d) = (col/3, col%3); cols 24..31 zero pad.
// ---------------------------------------------------------------------------
__global__ __launch_bounds__(256) void pack_x(const float* __restrict__ x,
                                              uint32_t* __restrict__ Xpk) {
  int widx = blockIdx.x * 256 + threadIdx.x;      // 64*2*256 = 32768
  if (widx >= 64 * 2 * 256) return;
  int v = widx & 7, lane = (widx >> 3) & 31;
  int ct = (widx >> 8) & 1, kc = widx >> 9;
  int col = ct * 16 + (lane & 15);
  uint32_t w = 0u;
  if (col < 24) {
    int b = col / 3, d = col - b * 3;
    int k = kc * 32 + frag_k(v, lane >> 4);
    const float* px = &x[((size_t)b * 2048 + k) * 3 + d];
    w = pack_bf2(px[0], px[3]);
  }
  Xpk[widx] = w;
}

// ---------------------------------------------------------------------------
// k1: xc[row=g*8+m][col=b*3+d] = sum_n W_map[row][n] * x[b][n][d]
// 128 blocks x 8 waves: waves 0-3 do K[0:1024), waves 4-7 do K[1024:2048)
// for row-tiles blk*4+0..3; LDS reduce; 4 accumulators spread WMMA hazards.
// ---------------------------------------------------------------------------
__global__ __launch_bounds__(256) void map_xc(const float* __restrict__ Wmap,
                                              const uint32_t* __restrict__ Xpk,
                                              float* __restrict__ xc) {
  __shared__ float red[4][16][32];                // 8 KB reduction buffer
  int wave = threadIdx.x >> 5, lane = threadIdx.x & 31;
  int rt   = blockIdx.x * 4 + (wave & 3);         // 0..511
  int kba  = (wave >> 2) * 32;                    // k-chunk base (0 or 32)
  int rlo  = lane & 15, half = lane >> 4;
  int row  = rt * 16 + rlo;
  const float* wrow = &Wmap[(size_t)row * 2048];
  const uint4* xb = (const uint4*)&Xpk[lane * 8]; // + (kc*2+ct)*64 uint4
  v8f a0a = {}, a0b = {}, a1a = {}, a1b = {};
  for (int kc = kba; kc < kba + 32; kc += 2) {
    FragBF A0, A1, B00, B01, B10, B11;
#pragma unroll
    for (int v = 0; v < 8; ++v) {
      int k = frag_k(v, half);
      A0.u[v] = pack_bf2(wrow[(kc + 0) * 32 + k], wrow[(kc + 0) * 32 + k + 1]);
      A1.u[v] = pack_bf2(wrow[(kc + 1) * 32 + k], wrow[(kc + 1) * 32 + k + 1]);
    }
    const uint4* p = xb + (size_t)kc * 128;       // 2 ct * 64 uint4 per kc
    B00.q[0] = p[0];   B00.q[1] = p[1];
    B01.q[0] = p[64];  B01.q[1] = p[65];
    B10.q[0] = p[128]; B10.q[1] = p[129];
    B11.q[0] = p[192]; B11.q[1] = p[193];
    a0a = WMMA_BF16(A0.f, B00.f, a0a);
    a1a = WMMA_BF16(A0.f, B01.f, a1a);
    a0b = WMMA_BF16(A1.f, B10.f, a0b);
    a1b = WMMA_BF16(A1.f, B11.f, a1b);
  }
  // K-split reduction: waves 4-7 dump partials, waves 0-3 combine + store.
  if (wave >= 4) {
#pragma unroll
    for (int r = 0; r < 8; ++r) {
      red[wave - 4][r][lane]     = a0a[r] + a0b[r];
      red[wave - 4][r + 8][lane] = a1a[r] + a1b[r];
    }
  }
  __syncthreads();
  if (wave < 4) {
    int colB1 = 16 + rlo;                         // cols 24..31 padding
#pragma unroll
    for (int r = 0; r < 8; ++r) {
      float c0 = a0a[r] + a0b[r] + red[wave][r][lane];
      float c1 = a1a[r] + a1b[r] + red[wave][r + 8][lane];
      int orow = rt * 16 + r + 8 * half;
      xc[orow * 24 + rlo] = c0;
      if (colB1 < 24) xc[orow * 24 + colB1] = c1;
    }
  }
}

// ---------------------------------------------------------------------------
// k2: fused basis construction + dual projection + att.
// ---------------------------------------------------------------------------
__global__ __launch_bounds__(256) void fused_att(const uint32_t* __restrict__ Apk,
                                                 const float* __restrict__ xc,
                                                 float* __restrict__ att) {
  // 160 KB LDS total -> 2 workgroups per 320KB WGP.
  __shared__ union {
    uint32_t bw[2][3 * SRCH * 2 * 256];           // 2 x 61440 B staged B panel
    float    xcs[BGTILE][8][3];                   // phase-A only (aliased)
  } sb;
  __shared__ float  s_delta[BGTILE][3][64];       // 24 KB
  __shared__ float2 s_pe[BGTILE][64];             // 16 KB {pref, exp(-dn)}

  int wg = blockIdx.x;                            // 256 workgroups
  int b  = wg >> 5;
  int g0 = (wg & 31) * BGTILE;
  int tid = threadIdx.x;

  // ---- Phase A0: xc tile (256 rows x 3 cols, row stride 24) -> LDS.
#if HAVE_TDM
  if (tid < 32) {                                 // wave 0 issues the DMA
    uint64_t ga = (uint64_t)(uintptr_t)&xc[((size_t)g0 * 8) * 24 + b * 3];
    uint32_t la = (uint32_t)(uintptr_t)&sb.xcs[0][0][0];   // LDS byte offset
    v4u d0 = { 1u,                                // count=1, no gather
               la,                                // lds_addr
               (uint32_t)ga,                      // global_addr[31:0]
               (uint32_t)((ga >> 32) & 0x01FFFFFFu) | 0x80000000u }; // +type=2
    v8i d1 = { (int)0x20000,                      // data_size=4B, mask=0
               (int)(3u << 16),                   // tensor_dim0 = 3
               (int)(256u << 16),                 // tensor_dim1 = 256
               (int)(3u << 16),                   // tile_dim0 = 3
               (int)256,                          // tile_dim1 = 256
               (int)24,                           // tensor_dim0_stride = 24
               0, 0 };
    v4i zz4 = { 0, 0, 0, 0 };
    v8i zz8 = { 0, 0, 0, 0, 0, 0, 0, 0 };
    __builtin_amdgcn_tensor_load_to_lds(d0, d1, zz4, zz4, zz8, 0);
    __builtin_amdgcn_s_wait_tensorcnt(0);
  }
#else
  for (int e = tid; e < BGTILE * 24; e += 256) {
    int bg = e / 24, rem = e - bg * 24, m = rem / 3, d = rem - m * 3;
    sb.xcs[bg][m][d] = xc[((g0 + bg) * 8 + m) * 24 + (b * 3 + d)];
  }
#endif
  __syncthreads();

  // ---- Phase A1: delta, pref = cutoff/(dn+eps)^2, E = exp(-dn)
  const float PI_OVER_CUT = 3.14159265358979f / 5.0f;
  for (int e = tid; e < BGTILE * 64; e += 256) {
    int p = e & 63, bg = e >> 6;
    int i = p >> 3, j = p & 7;                    // delta = xc[j] - xc[i]
    float dx = sb.xcs[bg][j][0] - sb.xcs[bg][i][0];
    float dy = sb.xcs[bg][j][1] - sb.xcs[bg][i][1];
    float dz = sb.xcs[bg][j][2] - sb.xcs[bg][i][2];
    float dn = sqrtf(dx * dx + dy * dy + dz * dz + 1e-5f);
    float cut = (dn < 5.0f) ? 0.5f * (__cosf(dn * PI_OVER_CUT) + 1.0f) : 0.0f;
    float inv = dn + 1e-5f;
    s_delta[bg][0][p] = dx;
    s_delta[bg][1][p] = dy;
    s_delta[bg][2][p] = dz;
    s_pe[bg][p] = make_float2(cut / (inv * inv), __expf(-dn));
  }
  __syncthreads();

  const float MEAN0 = 0.006737946999f;            // exp(-5)
  const float DMEAN = (1.0f - MEAN0) / 49.0f;
  const float TT    = (2.0f / 50.0f) * (1.0f - MEAN0);
  const float BETA  = 1.0f / (TT * TT);

  // Per-thread invariants for staging: word index == tid within each 256-blk.
  int svv   = tid & 7;                            // fragment VGPR slot
  int sln   = (tid >> 3) & 31;                    // fragment lane
  int scolA = sln & 15;
  int skf   = frag_k(svv, sln >> 4);              // even k within chunk

  // Cooperative staging of bf16 basis B-fragments (pre-swizzled word order).
  // exp computed once per (f,bg); d innermost shares sm across delta dims.
  auto stage = [&](int buf, int phase) {
#pragma unroll 1
    for (int chunk = 0; chunk < SRCH; ++chunk) {
      int k0 = phase * (SRCH * 32) + chunk * 32 + skf;
      int r = k0 >> 6, p = k0 & 63;               // f' = r*64 + p, p even
      float mean = MEAN0 + DMEAN * (float)r;
#pragma unroll
      for (int ct = 0; ct < 2; ++ct) {
        int col = ct * 16 + scolA;                // bg column
        float2 pe0 = s_pe[col][p];
        float2 pe1 = s_pe[col][p + 1];
        float t0 = pe0.y - mean, t1 = pe1.y - mean;
        float sm0 = pe0.x * __expf(-BETA * t0 * t0);
        float sm1 = pe1.x * __expf(-BETA * t1 * t1);
        int base = (chunk * 2 + ct) * 256 + tid;
#pragma unroll
        for (int d = 0; d < 3; ++d) {
          float dd0 = s_delta[col][d][p];
          float dd1 = s_delta[col][d][p + 1];
          sb.bw[buf][base + d * (SRCH * 2 * 256)] = pack_bf2(sm0 * dd0, sm1 * dd1);
        }
      }
    }
  };

  int wave = tid >> 5, lane = tid & 31;
  int rtL = wave, rtR = wave + 8;                 // left rows h, right rows h+128
  v8f accL[3][2] = {{{}, {}}, {{}, {}}, {{}, {}}};
  v8f accR[3][2] = {{{}, {}}, {{}, {}}, {{}, {}}};

  const uint4* paL = (const uint4*)&Apk[(size_t)(rtL * KCHUNKS) * 256 + lane * 8];
  const uint4* paR = (const uint4*)&Apk[(size_t)(rtR * KCHUNKS) * 256 + lane * 8];

  stage(0, 0);
  __syncthreads();
  for (int ph = 0; ph < KCHUNKS / SRCH; ++ph) {
    if (ph + 1 < KCHUNKS / SRCH) stage((ph + 1) & 1, ph + 1);
    int buf = ph & 1;
#pragma unroll 1
    for (int c = 0; c < SRCH; ++c) {
      int kc = ph * SRCH + c;
      __builtin_prefetch(paL + (size_t)(kc + 1) * 64, 0, 1);  // global_prefetch
      __builtin_prefetch(paR + (size_t)(kc + 1) * 64, 0, 1);
      FragBF a0, a1, bb;
      const uint4* p0 = paL + (size_t)kc * 64;
      a0.q[0] = p0[0]; a0.q[1] = p0[1];
      const uint4* p1 = paR + (size_t)kc * 64;
      a1.q[0] = p1[0]; a1.q[1] = p1[1];
#pragma unroll
      for (int d = 0; d < 3; ++d) {
#pragma unroll
        for (int ct = 0; ct < 2; ++ct) {
          const uint32_t* pb =
              &sb.bw[buf][((d * SRCH + c) * 2 + ct) * 256 + lane * 8];
          bb.q[0] = *(const uint4*)pb;
          bb.q[1] = *(const uint4*)(pb + 4);
          accL[d][ct] = WMMA_BF16(a0.f, bb.f, accL[d][ct]);
          accR[d][ct] = WMMA_BF16(a1.f, bb.f, accR[d][ct]);
        }
      }
    }
    __syncthreads();
  }

  // Lane-local epilogue: att[b,g,h] = sum_d left*right + 1e-5.
#pragma unroll
  for (int ct = 0; ct < 2; ++ct) {
    int col = ct * 16 + (lane & 15);
    int g = g0 + col;
    int h0 = wave * 16 + 8 * (lane >> 4);
    float outv[8];
#pragma unroll
    for (int r = 0; r < 8; ++r)
      outv[r] = accL[0][ct][r] * accR[0][ct][r] +
                accL[1][ct][r] * accR[1][ct][r] +
                accL[2][ct][r] * accR[2][ct][r] + 1e-5f;
    float4* dst = (float4*)&att[((size_t)b * 1024 + g) * 128 + h0];
    dst[0] = make_float4(outv[0], outv[1], outv[2], outv[3]);
    dst[1] = make_float4(outv[4], outv[5], outv[6], outv[7]);
  }
}

// ---------------------------------------------------------------------------
// k3: logsumexp over g, then silu MLP head.
// ---------------------------------------------------------------------------
__global__ __launch_bounds__(128) void pool_mlp(const float* __restrict__ att,
                                                const float* __restrict__ fc1w,
                                                const float* __restrict__ fc1b,
                                                const float* __restrict__ fc2w,
                                                const float* __restrict__ fc2b,
                                                float* __restrict__ out) {
  int b = blockIdx.x, h = threadIdx.x;
  const float* ab = &att[(size_t)b * 1024 * 128];
  float mx = -3.0e38f;
  for (int g = 0; g < 1024; ++g) mx = fmaxf(mx, ab[g * 128 + h]);
  float s = 0.0f;
  for (int g = 0; g < 1024; ++g) s += __expf(ab[g * 128 + h] - mx);
  float pool = __logf(s) + mx;
  __shared__ float sp[128], sh[128];
  sp[h] = pool;
  __syncthreads();
  float acc = fc1b[h];
  for (int k = 0; k < 128; ++k) acc += sp[k] * fc1w[h * 128 + k];
  sh[h] = acc / (1.0f + __expf(-acc));            // silu
  __syncthreads();
  if (h == 0) {
    float o = fc2b[0];
    for (int k = 0; k < 128; ++k) o += sh[k] * fc2w[k];
    out[b] = o;
  }
}

// ---------------------------------------------------------------------------
extern "C" void kernel_launch(void* const* d_in, const int* in_sizes, int n_in,
                              void* d_out, int out_size, void* d_ws, size_t ws_size,
                              hipStream_t stream) {
  const float* x    = (const float*)d_in[0];
  const float* Wmap = (const float*)d_in[1];
  const float* Wl   = (const float*)d_in[2];
  const float* Wr   = (const float*)d_in[3];
  const float* fc1w = (const float*)d_in[4];
  const float* fc1b = (const float*)d_in[5];
  const float* fc2w = (const float*)d_in[6];
  const float* fc2b = (const float*)d_in[7];

  char* ws = (char*)d_ws;
  uint32_t* Apk = (uint32_t*)ws;                        // 1,638,400 B
  uint32_t* Xpk = (uint32_t*)(ws + 1638400);            //   131,072 B
  float*    xcb = (float*)(ws + 1638400 + 131072);      //   786,432 B
  float*    att = (float*)(ws + 1638400 + 131072 + 786432);  // 4,194,304 B
  float*    out = (float*)d_out;

  pack_w   <<<1600, 256, 0, stream>>>(Wl, Wr, Apk);
  pack_x   <<<128,  256, 0, stream>>>(x, Xpk);
  map_xc   <<<128,  256, 0, stream>>>(Wmap, Xpk, xcb);
  fused_att<<<256,  256, 0, stream>>>(Apk, xcb, att);
  pool_mlp <<<8,    128, 0, stream>>>(att, fc1w, fc1b, fc2w, fc2b, out);
}
